// MOELayer_30236569764393
// MI455X (gfx1250) — matrix operand order repaired and checked by reading, not compiled, not run on hardware
//
#include <hip/hip_runtime.h>
#include <hip/hip_bf16.h>

typedef __attribute__((ext_vector_type(4)))  _Float16 v4h;
typedef __attribute__((ext_vector_type(8)))  _Float16 v8h;
typedef __attribute__((ext_vector_type(16))) _Float16 v16h;
typedef __attribute__((ext_vector_type(8)))  float    v8f;

#define T_TOK 8192
#define DIM   1024
#define NEXP  8

// ---------------- convert x (f32 -> f16), 4 elems/thread ----------------
__global__ __launch_bounds__(256) void cvt_x_kernel(const float* __restrict__ in,
                                                    _Float16* __restrict__ out, int n4) {
    int i = blockIdx.x * blockDim.x + threadIdx.x;
    if (i < n4) {
        float4 v = ((const float4*)in)[i];
        v4h h = {(_Float16)v.x, (_Float16)v.y, (_Float16)v.z, (_Float16)v.w};
        ((v4h*)out)[i] = h;
    }
}

// -------- transpose+convert We[e][d][f] (f32) -> WehT[e][f][d] (f16) --------
__global__ __launch_bounds__(256) void cvt_we_kernel(const float* __restrict__ we,
                                                     _Float16* __restrict__ wehT) {
    __shared__ float tile[32][33];
    int e  = blockIdx.z;
    int d0 = blockIdx.y * 32;
    int f0 = blockIdx.x * 32;
    int tx = threadIdx.x & 31;
    int ty = threadIdx.x >> 5;            // 0..7
    const float* src = we + ((size_t)e * DIM) * DIM;
#pragma unroll
    for (int i = 0; i < 4; ++i) {
        int d = d0 + ty + i * 8;
        tile[ty + i * 8][tx] = src[(size_t)d * DIM + f0 + tx];
    }
    __syncthreads();
    _Float16* dst = wehT + ((size_t)e * DIM) * DIM;
#pragma unroll
    for (int i = 0; i < 4; ++i) {
        int f = f0 + ty + i * 8;
        dst[(size_t)f * DIM + d0 + tx] = (_Float16)tile[tx][ty + i * 8];
    }
}

// ---------------- gating: one wave per token ----------------
__global__ __launch_bounds__(256) void gate_kernel(const float* __restrict__ x,
                                                   const float* __restrict__ wg,
                                                   int* __restrict__ cnt,
                                                   int* __restrict__ tok,
                                                   float* __restrict__ wgt) {
    int w    = threadIdx.x >> 5;
    int lane = threadIdx.x & 31;
    int t    = blockIdx.x * 8 + w;

    float acc[NEXP];
#pragma unroll
    for (int e = 0; e < NEXP; ++e) acc[e] = 0.f;

    const float4* wg4 = (const float4*)wg;
    for (int d = lane; d < DIM; d += 32) {
        float xv = x[(size_t)t * DIM + d];
        float4 g0 = wg4[d * 2 + 0];
        float4 g1 = wg4[d * 2 + 1];
        acc[0] += xv * g0.x; acc[1] += xv * g0.y;
        acc[2] += xv * g0.z; acc[3] += xv * g0.w;
        acc[4] += xv * g1.x; acc[5] += xv * g1.y;
        acc[6] += xv * g1.z; acc[7] += xv * g1.w;
    }
#pragma unroll
    for (int off = 16; off > 0; off >>= 1)
#pragma unroll
        for (int e = 0; e < NEXP; ++e) acc[e] += __shfl_xor(acc[e], off, 32);

    if (lane == 0) {
        int i0 = 0;
#pragma unroll
        for (int e = 1; e < NEXP; ++e) if (acc[e] > acc[i0]) i0 = e;
        int i1 = (i0 == 0) ? 1 : 0;
#pragma unroll
        for (int e = 0; e < NEXP; ++e) if (e != i0 && acc[e] > acc[i1]) i1 = e;
        float p1 = expf(acc[i1] - acc[i0]);   // l0 >= l1; softmax denom cancels
        float s  = 1.f / (1.f + p1);
        float w0 = s, w1 = p1 * s;
        int p = atomicAdd(&cnt[i0], 1);
        tok[i0 * T_TOK + p] = t; wgt[i0 * T_TOK + p] = w0;
        p = atomicAdd(&cnt[i1], 1);
        tok[i1 * T_TOK + p] = t; wgt[i1 * T_TOK + p] = w1;
    }
}

// ---------------- expert GEMM: gathered tokens, WMMA f16 -> f32 ----------------
__global__ __launch_bounds__(256) void moe_gemm_kernel(const _Float16* __restrict__ xh,
                                                       const _Float16* __restrict__ wehT,
                                                       const float* __restrict__ be,
                                                       const int* __restrict__ cnt,
                                                       const int* __restrict__ tok,
                                                       const float* __restrict__ wgt,
                                                       float* __restrict__ y) {
    int e  = blockIdx.z;
    int n0 = blockIdx.x * 128;
    int m0 = blockIdx.y * 128;
    int count = cnt[e];
    if (m0 >= count) return;

    int lane = threadIdx.x & 31;
    int w    = threadIdx.x >> 5;
    int wm   = w & 3;                 // 4 M-wave rows  -> 128 M per block
    int wn   = w >> 2;                // 2 N-wave cols  -> 128 N per block
    int mBase = m0 + wm * 32;
    int nBase = n0 + wn * 64;

    const int*   tokE = tok + e * T_TOK;
    const float* wgtE = wgt + e * T_TOK;

    int l16 = lane & 15;
    int hi  = lane >> 4;              // 0 / 1
    int khA = hi * 16;                // A: lanes16-31 hold K=16..31
    int khB = hi * 16;                // B: lanes16-31 hold K=16..31

    int row0 = mBase + l16;
    int row1 = mBase + 16 + l16;
    int t0 = (row0 < count) ? tokE[row0] : 0;
    int t1 = (row1 < count) ? tokE[row1] : 0;

    const _Float16* a0p = xh + (size_t)t0 * DIM + khA;
    const _Float16* a1p = xh + (size_t)t1 * DIM + khA;
    const _Float16* wb  = wehT + (size_t)e * DIM * DIM;
    const _Float16* bp0 = wb + (size_t)(nBase +  0 + l16) * DIM + khB;
    const _Float16* bp1 = wb + (size_t)(nBase + 16 + l16) * DIM + khB;
    const _Float16* bp2 = wb + (size_t)(nBase + 32 + l16) * DIM + khB;
    const _Float16* bp3 = wb + (size_t)(nBase + 48 + l16) * DIM + khB;

    v8f acc[2][4];
#pragma unroll
    for (int mi = 0; mi < 2; ++mi)
#pragma unroll
        for (int ni = 0; ni < 4; ++ni) acc[mi][ni] = (v8f)(0.f);

#pragma unroll 2
    for (int kb = 0; kb < DIM; kb += 32) {
        v8h a0lo = *(const v8h*)(a0p + kb);
        v8h a0hi = *(const v8h*)(a0p + kb + 8);
        v8h a1lo = *(const v8h*)(a1p + kb);
        v8h a1hi = *(const v8h*)(a1p + kb + 8);
        v16h a0 = __builtin_shufflevector(a0lo, a0hi, 0,1,2,3,4,5,6,7,8,9,10,11,12,13,14,15);
        v16h a1 = __builtin_shufflevector(a1lo, a1hi, 0,1,2,3,4,5,6,7,8,9,10,11,12,13,14,15);
        v16h b0 = *(const v16h*)(bp0 + kb);
        v16h b1 = *(const v16h*)(bp1 + kb);
        v16h b2 = *(const v16h*)(bp2 + kb);
        v16h b3 = *(const v16h*)(bp3 + kb);

        acc[0][0] = __builtin_amdgcn_wmma_f32_16x16x32_f16(false, a0, false, b0, (short)0, acc[0][0], false, false);
        acc[0][1] = __builtin_amdgcn_wmma_f32_16x16x32_f16(false, a0, false, b1, (short)0, acc[0][1], false, false);
        acc[0][2] = __builtin_amdgcn_wmma_f32_16x16x32_f16(false, a0, false, b2, (short)0, acc[0][2], false, false);
        acc[0][3] = __builtin_amdgcn_wmma_f32_16x16x32_f16(false, a0, false, b3, (short)0, acc[0][3], false, false);
        acc[1][0] = __builtin_amdgcn_wmma_f32_16x16x32_f16(false, a1, false, b0, (short)0, acc[1][0], false, false);
        acc[1][1] = __builtin_amdgcn_wmma_f32_16x16x32_f16(false, a1, false, b1, (short)0, acc[1][1], false, false);
        acc[1][2] = __builtin_amdgcn_wmma_f32_16x16x32_f16(false, a1, false, b2, (short)0, acc[1][2], false, false);
        acc[1][3] = __builtin_amdgcn_wmma_f32_16x16x32_f16(false, a1, false, b3, (short)0, acc[1][3], false, false);
    }

    // epilogue: bias, relu, gate weight, atomic combine
    int cRow = hi * 8;                // C/D: VGPR r holds M = r + (lane>>4)*8
#pragma unroll
    for (int ni = 0; ni < 4; ++ni) {
        int col = nBase + ni * 16 + l16;
        float bias = be[e * DIM + col];
#pragma unroll
        for (int mi = 0; mi < 2; ++mi) {
#pragma unroll
            for (int r = 0; r < 8; ++r) {
                int row = mBase + mi * 16 + cRow + r;
                if (row < count) {
                    float v = acc[mi][ni][r] + bias;
                    v = v > 0.f ? v : 0.f;
                    v *= wgtE[row];
                    atomicAdd(&y[(size_t)tokE[row] * DIM + col], v);
                }
            }
        }
    }
}

extern "C" void kernel_launch(void* const* d_in, const int* in_sizes, int n_in,
                              void* d_out, int out_size, void* d_ws, size_t ws_size,
                              hipStream_t stream) {
    const float* x  = (const float*)d_in[0];   // [T, D]
    const float* Wg = (const float*)d_in[1];   // [D, E]
    const float* We = (const float*)d_in[2];   // [E, D, D]
    const float* be = (const float*)d_in[3];   // [E, D]
    float* y = (float*)d_out;                  // [T, D]

    char* ws = (char*)d_ws;
    const size_t XH_BYTES  = (size_t)T_TOK * DIM * 2;        // 16 MB
    const size_t WEH_BYTES = (size_t)NEXP * DIM * DIM * 2;   // 16 MB
    _Float16* xh   = (_Float16*)ws;
    _Float16* wehT = (_Float16*)(ws + XH_BYTES);
    int*      cnt  = (int*)(ws + XH_BYTES + WEH_BYTES);
    int*      tok  = (int*)(ws + XH_BYTES + WEH_BYTES + 256);
    float*    wgt  = (float*)(ws + XH_BYTES + WEH_BYTES + 256 + (size_t)NEXP * T_TOK * 4);

    hipMemsetAsync(cnt, 0, NEXP * sizeof(int), stream);
    hipMemsetAsync(y, 0, (size_t)T_TOK * DIM * sizeof(float), stream);

    int n4 = T_TOK * DIM / 4;
    cvt_x_kernel<<<(n4 + 255) / 256, 256, 0, stream>>>(x, xh, n4);
    cvt_we_kernel<<<dim3(DIM / 32, DIM / 32, NEXP), 256, 0, stream>>>(We, wehT);
    gate_kernel<<<T_TOK / 8, 256, 0, stream>>>(x, Wg, cnt, tok, wgt);
    moe_gemm_kernel<<<dim3(DIM / 128, T_TOK / 128, NEXP), 256, 0, stream>>>(
        xh, wehT, be, cnt, tok, wgt, y);
}
